// GPT2SelfAttention_22514218566329
// MI455X (gfx1250) — matrix operand order repaired and checked by reading, not compiled, not run on hardware
//
#include <hip/hip_runtime.h>

typedef _Float16 h8   __attribute__((ext_vector_type(8)));
typedef _Float16 v16h __attribute__((ext_vector_type(16)));
typedef float    v8f  __attribute__((ext_vector_type(8)));

constexpr int Bn = 2, Sn = 2048, En = 1024, Hn = 16, Dn = 64;
constexpr int Mtot = Bn * Sn;            // 4096 rows for both GEMMs
constexpr float kScale = 0.125f;         // 1/sqrt(64)
constexpr float kNegBig = -3.0e38f;

constexpr int BM = 128, BN = 128, BK = 32, LDT = 40;  // LDS stride 40 f16 (80B)

// ---------- CDNA5 async global->LDS staging (ASYNCcnt path), with fallback ----
#if defined(__has_builtin)
#if __has_builtin(__builtin_amdgcn_global_load_async_to_lds_b128)
#define USE_ASYNC_LDS 1
#endif
#endif

#ifdef USE_ASYNC_LDS
// Parameter type leaked by the round-2 diagnostic: non-const pointer to
// 'int __attribute__((vector_size(16)))' in global AS; dest is the LDS AS.
typedef int v4i_vs __attribute__((vector_size(16)));
typedef __attribute__((address_space(1))) v4i_vs* gptr4;
typedef __attribute__((address_space(3))) v4i_vs* lptr4;
#endif

__device__ __forceinline__ void cp16(_Float16* dst, const _Float16* src) {
#ifdef USE_ASYNC_LDS
  __builtin_amdgcn_global_load_async_to_lds_b128((gptr4)src, (lptr4)dst, 0, 0);
#else
  *(h8*)dst = *(const h8*)src;
#endif
}

__device__ __forceinline__ void wait_stage() {
#ifdef USE_ASYNC_LDS
  asm volatile("s_wait_asynccnt 0x0" ::: "memory");
#endif
  __syncthreads();
}

// ---------- fragment helpers (CDNA5 WMMA VGPR layouts, wave32) ----------
__device__ __forceinline__ v16h frag2(const _Float16* p0, const _Float16* p1) {
  h8 a = *(const h8*)p0;
  h8 b = *(const h8*)p1;
  return __builtin_shufflevector(a, b, 0,1,2,3,4,5,6,7,8,9,10,11,12,13,14,15);
}

__device__ __forceinline__ v8f wmma16(v16h a, v16h b, v8f c) {
  return __builtin_amdgcn_wmma_f32_16x16x32_f16(false, a, false, b, (short)0, c,
                                                false, false);
}

// ---------- precision conversion kernels ----------
__global__ void k_f32_to_f16(const float* __restrict__ src,
                             _Float16* __restrict__ dst, int n) {
  int i = blockIdx.x * blockDim.x + threadIdx.x;
  if (i < n) dst[i] = (_Float16)src[i];
}

// wt[n][k] = (f16) w[k][n]   (w is [K][N] row-major, K == 1024 here)
__global__ void k_transpose_f16(const float* __restrict__ w,
                                _Float16* __restrict__ wt, int K, int N) {
  int i = blockIdx.x * blockDim.x + threadIdx.x;
  if (i < K * N) {
    int n = i / K;
    int k = i % K;
    wt[i] = (_Float16)w[(size_t)k * N + n];
  }
}

// ---------- double-buffered WMMA GEMM core: 128x128 block, K=1024 ----------
// 8 waves: 4 (M) x 2 (N); each wave owns 32x64 = 2x4 WMMA tiles.
__device__ __forceinline__ void stage(const _Float16* __restrict__ A,
                                      const _Float16* __restrict__ Bt,
                                      int m0, int n0, int k0,
                                      _Float16* lA, _Float16* lB, int tid) {
  int r = tid >> 2, cc = (tid & 3) * 8;        // 64 rows x 4 chunks per pass
  cp16(lA + r * LDT + cc,        A + (size_t)(m0 + r) * 1024 + k0 + cc);
  cp16(lA + (r + 64) * LDT + cc, A + (size_t)(m0 + r + 64) * 1024 + k0 + cc);
  cp16(lB + r * LDT + cc,        Bt + (size_t)(n0 + r) * 1024 + k0 + cc);
  cp16(lB + (r + 64) * LDT + cc, Bt + (size_t)(n0 + r + 64) * 1024 + k0 + cc);
}

__device__ __forceinline__ void gemm_core_k1024(const _Float16* __restrict__ A,
                                                const _Float16* __restrict__ Bt,
                                                int m0, int n0,
                                                _Float16* lA, _Float16* lB,
                                                v8f acc[2][4]) {
  const int tid = threadIdx.x;
  const int lane = tid & 31, l16 = lane & 15, hi = lane >> 4;
  const int w = tid >> 5, wm = w & 3, wn = w >> 2;
  const v8f zero = {0.f,0.f,0.f,0.f,0.f,0.f,0.f,0.f};
  for (int i = 0; i < 2; ++i)
    for (int j = 0; j < 4; ++j) acc[i][j] = zero;

  stage(A, Bt, m0, n0, 0, lA, lB, tid);        // prologue fill buffer 0
  wait_stage();

  for (int ks = 0; ks < 1024 / BK; ++ks) {
    const int cur = ks & 1;
    _Float16* cA = lA + cur * (BM * LDT);
    _Float16* cB = lB + cur * (BN * LDT);
    if (ks + 1 < 1024 / BK)                    // overlap next tile with WMMAs
      stage(A, Bt, m0, n0, (ks + 1) * BK,
            lA + (cur ^ 1) * (BM * LDT), lB + (cur ^ 1) * (BN * LDT), tid);

    v16h a[2], b[4];
#pragma unroll
    for (int i = 0; i < 2; ++i) {
      const _Float16* p = cA + (wm * 32 + i * 16 + l16) * LDT;
      a[i] = frag2(p + 8 * hi, p + 16 + 8 * hi);     // A-frag K map
    }
#pragma unroll
    for (int j = 0; j < 4; ++j) {
      const _Float16* p = cB + (wn * 64 + j * 16 + l16) * LDT;
      b[j] = frag2(p + 8 * hi, p + 16 + 8 * hi);     // B-frag K map
    }
#pragma unroll
    for (int i = 0; i < 2; ++i)
#pragma unroll
      for (int j = 0; j < 4; ++j) acc[i][j] = wmma16(a[i], b[j], acc[i][j]);

    wait_stage();                                    // one barrier per k-step
  }
}

// ---------- QKV projection: writes Q,K as [B,H,S,D] f16 and V^T as [B,H,D,S] --
__global__ __launch_bounds__(256) void k_qkv_gemm(
    const _Float16* __restrict__ xh, const _Float16* __restrict__ wt,
    const float* __restrict__ bias, _Float16* __restrict__ qb,
    _Float16* __restrict__ kb, _Float16* __restrict__ vt) {
  __shared__ _Float16 lA[2 * BM * LDT];
  __shared__ _Float16 lB[2 * BN * LDT];
  int m0 = blockIdx.y * BM, n0 = blockIdx.x * BN;
  v8f acc[2][4];
  gemm_core_k1024(xh, wt, m0, n0, lA, lB, acc);

  const int lane = threadIdx.x & 31, l16 = lane & 15, hi = lane >> 4;
  const int w = threadIdx.x >> 5, wm = w & 3, wn = w >> 2;
#pragma unroll
  for (int i = 0; i < 2; ++i)
#pragma unroll
    for (int j = 0; j < 4; ++j) {
      int gn = n0 + wn * 64 + j * 16 + l16;          // [0, 3072)
      float bv = bias[gn];
      int region = gn >> 10, col = gn & 1023;        // uniform per 16-wide tile
      int hh = col >> 6, d = col & 63;
      int gmb = m0 + wm * 32 + i * 16 + 8 * hi;
      if (region == 0) {
#pragma unroll
        for (int r = 0; r < 8; ++r) {
          int gm = gmb + r, b = gm >> 11, s = gm & 2047;
          qb[((size_t)(b * Hn + hh) * Sn + s) * Dn + d] =
              (_Float16)(acc[i][j][r] + bv);
        }
      } else if (region == 1) {
#pragma unroll
        for (int r = 0; r < 8; ++r) {
          int gm = gmb + r, b = gm >> 11, s = gm & 2047;
          kb[((size_t)(b * Hn + hh) * Sn + s) * Dn + d] =
              (_Float16)(acc[i][j][r] + bv);
        }
      } else {
#pragma unroll
        for (int r = 0; r < 8; ++r) {
          int gm = gmb + r, b = gm >> 11, s = gm & 2047;
          vt[((size_t)(b * Hn + hh) * Dn + d) * Sn + s] =
              (_Float16)(acc[i][j][r] + bv);
        }
      }
    }
}

// ---------- flash-style causal attention: one wave per 16-query tile ----------
__global__ __launch_bounds__(256) void k_flash_attn(
    const _Float16* __restrict__ qb, const _Float16* __restrict__ kb,
    const _Float16* __restrict__ vt, _Float16* __restrict__ ctx) {
  __shared__ _Float16 pl[8][16 * 32];   // per-wave P bounce buffer (1KB each)
  const int w = threadIdx.x >> 5;
  const int lane = threadIdx.x & 31, l16 = lane & 15, hi = lane >> 4;
  int t = blockIdx.x * 8 + w;           // 4096 q-tiles total
  int qt = t & 127;                     // S/16
  int bh = t >> 7;                      // b*H + h
  int b = bh >> 4, h = bh & 15;
  int q0 = qt * 16;
  const _Float16* Qh = qb + (size_t)bh * Sn * Dn;
  const _Float16* Kh = kb + (size_t)bh * Sn * Dn;
  const _Float16* Vh = vt + (size_t)bh * Dn * Sn;
  _Float16* plw = pl[w];

  v16h qa[2];
#pragma unroll
  for (int dh = 0; dh < 2; ++dh) {
    const _Float16* p = Qh + (size_t)(q0 + l16) * Dn + 32 * dh;
    qa[dh] = frag2(p + 8 * hi, p + 16 + 8 * hi);
  }

  const v8f zero = {0.f,0.f,0.f,0.f,0.f,0.f,0.f,0.f};
  float m_[8], ls[8];
  v8f o[4];
#pragma unroll
  for (int r = 0; r < 8; ++r) { m_[r] = kNegBig; ls[r] = 0.f; }
#pragma unroll
  for (int dn = 0; dn < 4; ++dn) o[dn] = zero;

  int nch = (q0 + 15) / 32 + 1;         // causal: only chunks with keys <= q0+15
  for (int c = 0; c < nch; ++c) {
    int kbase = c * 32;
    if (c + 1 < nch) {                  // prefetch next K/V chunk into caches
      __builtin_prefetch(Kh + (size_t)(kbase + 32 + lane) * Dn, 0, 1);
      __builtin_prefetch(Vh + (size_t)(lane * 2) * Sn + kbase + 32, 0, 1);
    }
    v8f s0 = zero, s1 = zero;
#pragma unroll
    for (int dh = 0; dh < 2; ++dh) {    // reduce over D=64 in two K=32 steps
      const _Float16* p0 = Kh + (size_t)(kbase + l16) * Dn + 32 * dh;
      const _Float16* p1 = Kh + (size_t)(kbase + 16 + l16) * Dn + 32 * dh;
      s0 = wmma16(qa[dh], frag2(p0 + 8 * hi, p0 + 16 + 8 * hi), s0);
      s1 = wmma16(qa[dh], frag2(p1 + 8 * hi, p1 + 16 + 8 * hi), s1);
    }
    int key0 = kbase + l16, key1 = kbase + 16 + l16;
    float mx[8];
#pragma unroll
    for (int r = 0; r < 8; ++r) {
      int qrow = q0 + r + 8 * hi;
      float a0 = (key0 <= qrow) ? s0[r] * kScale : kNegBig;
      float a1 = (key1 <= qrow) ? s1[r] * kScale : kNegBig;
      s0[r] = a0; s1[r] = a1;
      mx[r] = fmaxf(a0, a1);
    }
#pragma unroll
    for (int dsh = 1; dsh < 16; dsh <<= 1)
#pragma unroll
      for (int r = 0; r < 8; ++r)
        mx[r] = fmaxf(mx[r], __shfl_xor(mx[r], dsh, 32));

    float alpha[8], rs[8];
#pragma unroll
    for (int r = 0; r < 8; ++r) {
      float nm = fmaxf(m_[r], mx[r]);
      alpha[r] = __expf(m_[r] - nm);
      float p0v = __expf(s0[r] - nm);
      float p1v = __expf(s1[r] - nm);
      m_[r] = nm;
      rs[r] = p0v + p1v;
      plw[(r + 8 * hi) * 32 + l16]      = (_Float16)p0v;
      plw[(r + 8 * hi) * 32 + 16 + l16] = (_Float16)p1v;
    }
#pragma unroll
    for (int dsh = 1; dsh < 16; dsh <<= 1)
#pragma unroll
      for (int r = 0; r < 8; ++r) rs[r] += __shfl_xor(rs[r], dsh, 32);
#pragma unroll
    for (int r = 0; r < 8; ++r) ls[r] = ls[r] * alpha[r] + rs[r];
#pragma unroll
    for (int dn = 0; dn < 4; ++dn)
#pragma unroll
      for (int r = 0; r < 8; ++r) o[dn][r] *= alpha[r];

    asm volatile("s_wait_dscnt 0" ::: "memory");   // P stores visible to reload
    const _Float16* pp = plw + l16 * 32;
    v16h pa = frag2(pp + 8 * hi, pp + 16 + 8 * hi); // P as A-fragment (16x32)
#pragma unroll
    for (int dn = 0; dn < 4; ++dn) {
      const _Float16* vp = Vh + (size_t)(dn * 16 + l16) * Sn + kbase;
      o[dn] = wmma16(pa, frag2(vp + 8 * hi, vp + 16 + 8 * hi), o[dn]);
    }
  }

  float inv[8];
#pragma unroll
  for (int r = 0; r < 8; ++r) inv[r] = 1.0f / ls[r];
#pragma unroll
  for (int dn = 0; dn < 4; ++dn) {
    int colbase = h * Dn + dn * 16 + l16;
#pragma unroll
    for (int r = 0; r < 8; ++r) {
      int srow = q0 + r + 8 * hi;
      ctx[((size_t)b * Sn + srow) * En + colbase] = (_Float16)(o[dn][r] * inv[r]);
    }
  }
}

// ---------- output projection: f32 result ----------
__global__ __launch_bounds__(256) void k_proj_gemm(
    const _Float16* __restrict__ ctx, const _Float16* __restrict__ wt,
    const float* __restrict__ bias, float* __restrict__ out) {
  __shared__ _Float16 lA[2 * BM * LDT];
  __shared__ _Float16 lB[2 * BN * LDT];
  int m0 = blockIdx.y * BM, n0 = blockIdx.x * BN;
  v8f acc[2][4];
  gemm_core_k1024(ctx, wt, m0, n0, lA, lB, acc);

  const int lane = threadIdx.x & 31, l16 = lane & 15, hi = lane >> 4;
  const int w = threadIdx.x >> 5, wm = w & 3, wn = w >> 2;
#pragma unroll
  for (int i = 0; i < 2; ++i)
#pragma unroll
    for (int j = 0; j < 4; ++j) {
      int gn = n0 + wn * 64 + j * 16 + l16;
      float bv = bias[gn];
#pragma unroll
      for (int r = 0; r < 8; ++r) {
        int gm = m0 + wm * 32 + i * 16 + r + 8 * hi;
        out[(size_t)gm * 1024 + gn] = acc[i][j][r] + bv;
      }
    }
}

extern "C" void kernel_launch(void* const* d_in, const int* in_sizes, int n_in,
                              void* d_out, int out_size, void* d_ws, size_t ws_size,
                              hipStream_t stream) {
  (void)in_sizes; (void)n_in; (void)out_size; (void)ws_size;
  const float* x      = (const float*)d_in[0];
  const float* w_attn = (const float*)d_in[1];
  const float* b_attn = (const float*)d_in[2];
  const float* w_proj = (const float*)d_in[3];
  const float* b_proj = (const float*)d_in[4];
  float* out = (float*)d_out;

  char* ws = (char*)d_ws;                       // 48 MiB layout
  _Float16* xh  = (_Float16*)(ws);              //  0 MiB: x f16        (8 MiB)
  _Float16* waT = (_Float16*)(ws + (size_t)8  * (1u << 20));  // w_attn^T (6 MiB)
  _Float16* wpT = (_Float16*)(ws + (size_t)14 * (1u << 20));  // w_proj^T (2 MiB)
  _Float16* qb  = (_Float16*)(ws + (size_t)16 * (1u << 20));  // Q [B,H,S,D]
  _Float16* kb  = (_Float16*)(ws + (size_t)24 * (1u << 20));  // K [B,H,S,D]
  _Float16* vt  = (_Float16*)(ws + (size_t)32 * (1u << 20));  // V^T [B,H,D,S]
  _Float16* cx  = (_Float16*)(ws + (size_t)40 * (1u << 20));  // merged ctx f16

  k_f32_to_f16<<<(Mtot * En) / 256, 256, 0, stream>>>(x, xh, Mtot * En);
  k_transpose_f16<<<(3 * En * En) / 256, 256, 0, stream>>>(w_attn, waT, En, 3 * En);
  k_transpose_f16<<<(En * En) / 256, 256, 0, stream>>>(w_proj, wpT, En, En);
  k_qkv_gemm<<<dim3(3072 / BN, Mtot / BM), 256, 0, stream>>>(xh, waT, b_attn,
                                                             qb, kb, vt);
  k_flash_attn<<<512, 256, 0, stream>>>(qb, kb, vt, cx);
  k_proj_gemm<<<dim3(1024 / BN, Mtot / BM), 256, 0, stream>>>(cx, wpT, b_proj, out);
}
